// GeneralTransformerBlock_18296560681612
// MI455X (gfx1250) — matrix operand order repaired and checked by reading, not compiled, not run on hardware
//
#include <hip/hip_runtime.h>

// ---------------- problem constants ----------------
#define B_    8
#define C_    512
#define HW_   3136
#define NH_   16
#define L_    49
#define HID_  2048
#define HD_   32
#define NWIN_ 512            // B * 8 * 8 windows
#define MTOT_ 25088          // NWIN_ * L_ = B*HW

typedef __attribute__((ext_vector_type(16))) _Float16 v16h;
typedef __attribute__((ext_vector_type(8)))  _Float16 v8h;
typedef __attribute__((ext_vector_type(8)))  float    v8f;
typedef unsigned int       u32;
typedef unsigned long long u64;
typedef __attribute__((ext_vector_type(4))) u32 u32x4;
typedef __attribute__((ext_vector_type(8))) int i32x8;
typedef __attribute__((ext_vector_type(4))) int i32x4;

#ifndef USE_TDM
#define USE_TDM 1
#endif
#if USE_TDM && __has_builtin(__builtin_amdgcn_tensor_load_to_lds) && \
    __has_builtin(__builtin_amdgcn_s_wait_tensorcnt)
#define HAVE_TDM 1
#else
#define HAVE_TDM 0
#endif

union Frag { v16h v; v8h h[2]; };

__device__ __forceinline__ v8f wmma_f16(const Frag& a, const Frag& b, v8f c) {
  // D = A(16x32 f16) x B(32x16 f16) + C(16x16 f32)
  return __builtin_amdgcn_wmma_f32_16x16x32_f16(false, a.v, false, b.v, (short)0, c, false, false);
}

__device__ __forceinline__ float gelu_exact(float x) {
  return 0.5f * x * (1.0f + erff(x * 0.70710678118654752f));
}

#if HAVE_TDM
__device__ __forceinline__ u32 lds_addr_of(const void* p) {
  return (u32)(unsigned long long)(__attribute__((address_space(3))) const void*)p;
}

// TDM: load a 128-row x 32-half tile (row stride = strideElems halfs in memory)
// into LDS with 8 halfs of padding after every row (LDS row stride 40 halfs).
__device__ __forceinline__ void tdm_load_tile_128x32(const _Float16* gsrc, u32 ldsAddr,
                                                     int strideElems) {
  u64 ga = (u64)(unsigned long long)gsrc;
  u32x4 g0;
  g0.x = 1u;                                               // count=1, user mode
  g0.y = ldsAddr;                                          // lds_addr (bytes)
  g0.z = (u32)ga;                                          // global_addr[31:0]
  g0.w = (u32)((ga >> 32) & 0x1FFFFFFull) | 0x80000000u;   // global_addr[56:32] | type=2
  i32x8 g1;
  // data_size=1 (2B), pad_enable, pad_interval=3 (16 dwords), pad_amount=3 (4 dwords)
  g1[0] = (int)((1u << 16) | (1u << 20) | (3u << 22) | (3u << 25));
  g1[1] = (int)0xFFFF0000u;                                // tensor_dim0 = 0x7FFFFFFF lo16
  g1[2] = (int)0xFFFF7FFFu;                                // tensor_dim0 hi | tensor_dim1 lo16
  g1[3] = (int)(0x7FFFu | (32u << 16));                    // tensor_dim1 hi | tile_dim0 = 32
  g1[4] = 128;                                             // tile_dim1 = 128, tile_dim2 = 0
  g1[5] = strideElems;                                     // tensor_dim0_stride (elements)
  g1[6] = 0;
  g1[7] = 0;
  i32x4 z4; z4[0] = 0; z4[1] = 0; z4[2] = 0; z4[3] = 0;
  i32x8 z8; z8[0] = 0; z8[1] = 0; z8[2] = 0; z8[3] = 0;
  z8[4] = 0; z8[5] = 0; z8[6] = 0; z8[7] = 0;
  __builtin_amdgcn_tensor_load_to_lds(g0, g1, z4, z4, z8, 0);
}
#endif

// ---------------- prep: BN folds + padded RPE bias expansion ----------------
__global__ void prep_kernel(const float* __restrict__ fc1_b,
                            const float* __restrict__ bn1_g, const float* __restrict__ bn1_b,
                            const float* __restrict__ bn1_m, const float* __restrict__ bn1_v,
                            const float* __restrict__ dw_b,
                            const float* __restrict__ bn2_g, const float* __restrict__ bn2_b,
                            const float* __restrict__ bn2_m, const float* __restrict__ bn2_v,
                            const float* __restrict__ fc2_b,
                            const float* __restrict__ bn3_g, const float* __restrict__ bn3_b,
                            const float* __restrict__ bn3_m, const float* __restrict__ bn3_v,
                            const float* __restrict__ rpb, const int* __restrict__ rel_index,
                            float* __restrict__ ab1, float* __restrict__ ab2,
                            float* __restrict__ ab3, float* __restrict__ biasPad) {
  int i = blockIdx.x * 256 + threadIdx.x;
  if (i < 2048) {
    float s = bn1_g[i] * rsqrtf(bn1_v[i] + 1e-5f);
    ab1[i] = s;
    ab1[2048 + i] = (fc1_b[i] - bn1_m[i]) * s + bn1_b[i];
  } else if (i < 4096) {
    int c = i - 2048;
    float s = bn2_g[c] * rsqrtf(bn2_v[c] + 1e-5f);
    ab2[c] = s;
    ab2[2048 + c] = (dw_b[c] - bn2_m[c]) * s + bn2_b[c];
  } else if (i < 4608) {
    int c = i - 4096;
    float s = bn3_g[c] * rsqrtf(bn3_v[c] + 1e-5f);
    ab3[c] = s;
    ab3[512 + c] = (fc2_b[c] - bn3_m[c]) * s + bn3_b[c];
  } else if (i < 4608 + NH_ * 64 * 64) {
    int j = i - 4608;
    int h = j >> 12;           // / 4096
    int row = (j >> 6) & 63;
    int col = j & 63;
    float v = -1e30f;
    if (row < L_ && col < L_) v = rpb[rel_index[row * L_ + col] * NH_ + h];
    biasPad[j] = v;            // [16][64][64], padded with -1e30
  }
}

// ---------------- f32 -> f16 conversion ----------------
__global__ void cvt_f32_f16(const float* __restrict__ s, _Float16* __restrict__ d, int n) {
  for (int i = blockIdx.x * blockDim.x + threadIdx.x; i < n; i += gridDim.x * blockDim.x)
    d[i] = (_Float16)s[i];
}

// ---------------- LN1 + window partition (one wave per token) ----------------
__global__ void __launch_bounds__(256)
ln1_win_kernel(const float* __restrict__ x, const float* __restrict__ g,
               const float* __restrict__ bta, _Float16* __restrict__ hA) {
  const int wave = threadIdx.x >> 5, lane = threadIdx.x & 31;
  const int token = blockIdx.x * 8 + wave;
  const int bb = token / HW_, n = token - bb * HW_;
  float vals[16];
  float s = 0.f, s2 = 0.f;
#pragma unroll
  for (int j = 0; j < 16; ++j) {
    int c = lane + 32 * j;
    float v = x[((size_t)bb * C_ + c) * HW_ + n];
    vals[j] = v; s += v; s2 += v * v;
  }
#pragma unroll
  for (int m = 16; m >= 1; m >>= 1) {
    s += __shfl_xor(s, m, 32);
    s2 += __shfl_xor(s2, m, 32);
  }
  float mean = s * (1.f / 512.f);
  float var = s2 * (1.f / 512.f) - mean * mean;
  float rs = rsqrtf(var + 1e-6f);
  int y = n / 56, xx = n - y * 56;
  int row = ((bb * 64) + (y / 7) * 8 + (xx / 7)) * L_ + (y % 7) * 7 + (xx % 7);
#pragma unroll
  for (int j = 0; j < 16; ++j) {
    int c = lane + 32 * j;
    hA[(size_t)row * C_ + c] = (_Float16)((vals[j] - mean) * rs * g[c] + bta[c]);
  }
}

// ---------------- LN2 (token-major in and out) ----------------
__global__ void __launch_bounds__(256)
ln2_kernel(const float* __restrict__ xs1, const float* __restrict__ g,
           const float* __restrict__ bta, _Float16* __restrict__ A2) {
  const int wave = threadIdx.x >> 5, lane = threadIdx.x & 31;
  const int token = blockIdx.x * 8 + wave;
  float vals[16];
  float s = 0.f, s2 = 0.f;
#pragma unroll
  for (int j = 0; j < 16; ++j) {
    int c = lane + 32 * j;
    float v = xs1[(size_t)token * C_ + c];
    vals[j] = v; s += v; s2 += v * v;
  }
#pragma unroll
  for (int m = 16; m >= 1; m >>= 1) {
    s += __shfl_xor(s, m, 32);
    s2 += __shfl_xor(s2, m, 32);
  }
  float mean = s * (1.f / 512.f);
  float var = s2 * (1.f / 512.f) - mean * mean;
  float rs = rsqrtf(var + 1e-6f);
#pragma unroll
  for (int j = 0; j < 16; ++j) {
    int c = lane + 32 * j;
    A2[(size_t)token * C_ + c] = (_Float16)((vals[j] - mean) * rs * g[c] + bta[c]);
  }
}

// ---------------- tiled WMMA GEMM: C = A[M,K] * Bt[N,K]^T, + epilogue ----------------
#define BM 128
#define BN 128
#define BK 32
#define LDST 40   // padded LDS row stride (halfs): 20 dwords -> conflict-free frag reads

enum { EPI_QKV = 0, EPI_PROJ_RES = 1, EPI_BN_GELU = 2, EPI_FC2_OUT = 3 };

template <int EPI>
__global__ void __launch_bounds__(256)
gemm_wmma_epi(const _Float16* __restrict__ A, const _Float16* __restrict__ Bt,
              int M, int N, int K,
              const float* __restrict__ e0, const float* __restrict__ e1,
              const float* __restrict__ e2, void* __restrict__ OutP) {
  __shared__ __attribute__((aligned(128))) _Float16 sA[2][BM * LDST];
  __shared__ __attribute__((aligned(128))) _Float16 sB[2][BN * LDST];

  const int t = threadIdx.x;
  const int bm0 = blockIdx.x * BM;
  const int bn0 = blockIdx.y * BN;
  const int wave = t >> 5, lane = t & 31;
  const int hi = lane >> 4, lr = lane & 15;
  const int waveM0 = (wave >> 2) * 64;   // 2 waves along M
  const int waveN0 = (wave & 3) * 32;    // 4 waves along N

#if HAVE_TDM
  // wave 0 drives the Tensor Data Mover for both tiles of k-step 0
  if (wave == 0) {
    tdm_load_tile_128x32(A + (size_t)bm0 * K, lds_addr_of(&sA[0][0]), K);
    tdm_load_tile_128x32(Bt + (size_t)bn0 * K, lds_addr_of(&sB[0][0]), K);
    __builtin_amdgcn_s_wait_tensorcnt(0);
  }
  __syncthreads();
#else
  const int rA0 = t >> 2, pA0 = t & 3;
  const int rA1 = (t + 256) >> 2, pA1 = (t + 256) & 3;
  float4 ra0 = *(const float4*)(A + (size_t)(bm0 + rA0) * K + pA0 * 8);
  float4 ra1 = *(const float4*)(A + (size_t)(bm0 + rA1) * K + pA1 * 8);
  float4 rb0 = *(const float4*)(Bt + (size_t)(bn0 + rA0) * K + pA0 * 8);
  float4 rb1 = *(const float4*)(Bt + (size_t)(bn0 + rA1) * K + pA1 * 8);
  *(float4*)(&sA[0][rA0 * LDST + pA0 * 8]) = ra0;
  *(float4*)(&sA[0][rA1 * LDST + pA1 * 8]) = ra1;
  *(float4*)(&sB[0][rA0 * LDST + pA0 * 8]) = rb0;
  *(float4*)(&sB[0][rA1 * LDST + pA1 * 8]) = rb1;
  __syncthreads();
#endif

  v8f acc[4][2];
#pragma unroll
  for (int mt = 0; mt < 4; ++mt)
#pragma unroll
    for (int nt = 0; nt < 2; ++nt) {
      v8f z = {0.f, 0.f, 0.f, 0.f, 0.f, 0.f, 0.f, 0.f};
      acc[mt][nt] = z;
    }

  const int ksteps = K / BK;
  for (int ks = 0; ks < ksteps; ++ks) {
    const int buf = ks & 1;
    const bool hasNext = (ks + 1 < ksteps);
#if HAVE_TDM
    if (hasNext && wave == 0) {   // overlap next-tile DMA with this step's WMMAs
      const int k0n = (ks + 1) * BK;
      tdm_load_tile_128x32(A + (size_t)bm0 * K + k0n, lds_addr_of(&sA[buf ^ 1][0]), K);
      tdm_load_tile_128x32(Bt + (size_t)bn0 * K + k0n, lds_addr_of(&sB[buf ^ 1][0]), K);
    }
#else
    if (hasNext) {
      const int k0n = (ks + 1) * BK;
      ra0 = *(const float4*)(A + (size_t)(bm0 + rA0) * K + k0n + pA0 * 8);
      ra1 = *(const float4*)(A + (size_t)(bm0 + rA1) * K + k0n + pA1 * 8);
      rb0 = *(const float4*)(Bt + (size_t)(bn0 + rA0) * K + k0n + pA0 * 8);
      rb1 = *(const float4*)(Bt + (size_t)(bn0 + rA1) * K + k0n + pA1 * 8);
      if (ks + 2 < ksteps) {
        __builtin_prefetch(A + (size_t)(bm0 + rA0) * K + (ks + 2) * BK + pA0 * 8, 0, 1);
        __builtin_prefetch(Bt + (size_t)(bn0 + rA0) * K + (ks + 2) * BK + pA0 * 8, 0, 1);
      }
    }
#endif

    Frag af[4];
#pragma unroll
    for (int mt = 0; mt < 4; ++mt) {
      const _Float16* base = &sA[buf][(waveM0 + 16 * mt + lr) * LDST];
      af[mt].h[0] = *(const v8h*)(base + hi * 8);
      af[mt].h[1] = *(const v8h*)(base + 16 + hi * 8);
    }
    Frag bf[2];
#pragma unroll
    for (int nt = 0; nt < 2; ++nt) {
      const _Float16* base = &sB[buf][(waveN0 + 16 * nt + lr) * LDST];
      bf[nt].h[0] = *(const v8h*)(base + hi * 8);
      bf[nt].h[1] = *(const v8h*)(base + 16 + hi * 8);
    }
#pragma unroll
    for (int mt = 0; mt < 4; ++mt)
#pragma unroll
      for (int nt = 0; nt < 2; ++nt)
        acc[mt][nt] = wmma_f16(af[mt], bf[nt], acc[mt][nt]);

#if HAVE_TDM
    if (hasNext && wave == 0) __builtin_amdgcn_s_wait_tensorcnt(0);
#else
    if (hasNext) {
      const int nb = buf ^ 1;
      *(float4*)(&sA[nb][rA0 * LDST + pA0 * 8]) = ra0;
      *(float4*)(&sA[nb][rA1 * LDST + pA1 * 8]) = ra1;
      *(float4*)(&sB[nb][rA0 * LDST + pA0 * 8]) = rb0;
      *(float4*)(&sB[nb][rA1 * LDST + pA1 * 8]) = rb1;
    }
#endif
    __syncthreads();
  }

  // -------- epilogue --------
#pragma unroll
  for (int mt = 0; mt < 4; ++mt)
#pragma unroll
    for (int nt = 0; nt < 2; ++nt)
#pragma unroll
      for (int r = 0; r < 8; ++r) {
        float val = acc[mt][nt][r];
        int gM = bm0 + waveM0 + mt * 16 + hi * 8 + r;
        int gN = bn0 + waveN0 + nt * 16 + lr;
        if constexpr (EPI == EPI_QKV) {
          float v = val + e0[gN];
          int which = gN >> 9, hh = (gN >> 5) & 15, d = gN & 31;
          if (which == 0) v *= 0.17677669529663687f;  // HD^-0.5
          int w = gM / L_, l = gM - w * L_;
          ((_Float16*)OutP)[(((size_t)which * NWIN_ + w) * NH_ + hh) * (L_ * HD_) + l * HD_ + d] =
              (_Float16)v;
        } else if constexpr (EPI == EPI_PROJ_RES) {
          int w = gM / L_, l = gM - w * L_;
          int bb = w >> 6, wi = w & 63;
          int y = (wi >> 3) * 7 + l / 7, xx = (wi & 7) * 7 + l % 7;
          int n = y * 56 + xx;
          float res = e2[((size_t)bb * C_ + gN) * HW_ + n];  // x[b][c][n]
          ((float*)OutP)[((size_t)bb * HW_ + n) * C_ + gN] = res + val + e0[gN];
        } else if constexpr (EPI == EPI_BN_GELU) {
          float y = val * e0[gN] + e1[gN];
          ((_Float16*)OutP)[(size_t)gM * N + gN] = (_Float16)gelu_exact(y);
        } else {  // EPI_FC2_OUT
          float y = val * e0[gN] + e1[gN];
          float g = gelu_exact(y);
          int bb = gM / HW_, n = gM - bb * HW_;
          float res = e2[((size_t)bb * HW_ + n) * C_ + gN];  // xs1[b][n][c]
          ((float*)OutP)[((size_t)bb * C_ + gN) * HW_ + n] = res + g;
        }
      }
}

// ---------------- windowed attention: one wave per (window, head) ----------------
__global__ void __launch_bounds__(32)
attn_kernel(const _Float16* __restrict__ qkv, const float* __restrict__ biasPad,
            _Float16* __restrict__ Ao) {
  const int wh = blockIdx.x;          // 0..8191
  const int w = wh >> 4, h = wh & 15;
  const int lane = threadIdx.x;
  const int hi = lane >> 4, lr = lane & 15;

  const size_t blk = (size_t)(w * NH_ + h) * (L_ * HD_);
  const _Float16* qp = qkv + blk;
  const _Float16* kp = qkv + (size_t)NWIN_ * NH_ * (L_ * HD_) + blk;
  const _Float16* vp = qkv + (size_t)2 * NWIN_ * NH_ * (L_ * HD_) + blk;
  const float* bp = biasPad + (size_t)h * 64 * 64;

  __shared__ __attribute__((aligned(128))) _Float16 sVt[32][72];  // v^T, padded to 64 cols
  __shared__ __attribute__((aligned(128))) _Float16 sP[16][72];   // softmax strip

  // stage v transposed (pad rows >=49 with zero)
  for (int i = lane; i < 32 * 64; i += 32) {
    int m = i >> 5, d = i & 31;
    sVt[d][m] = (m < L_) ? vp[m * HD_ + d] : (_Float16)0.f;
  }

  // k as B-fragments for S = q k^T (4 N-tiles covering cols 0..63)
  Frag kb[4];
#pragma unroll
  for (int nt = 0; nt < 4; ++nt) {
    const _Float16* base = kp + (16 * nt + lr) * HD_;
    kb[nt].h[0] = *(const v8h*)(base + hi * 8);
    kb[nt].h[1] = *(const v8h*)(base + 16 + hi * 8);
  }

  for (int mt = 0; mt < 4; ++mt) {
    // q A-fragment for this 16-row strip
    Frag qa;
    {
      const _Float16* base = qp + (16 * mt + lr) * HD_;
      qa.h[0] = *(const v8h*)(base + hi * 8);
      qa.h[1] = *(const v8h*)(base + 16 + hi * 8);
    }
    v8f sAcc[4];
#pragma unroll
    for (int nt = 0; nt < 4; ++nt) {
      v8f z = {0.f, 0.f, 0.f, 0.f, 0.f, 0.f, 0.f, 0.f};
      sAcc[nt] = z;
      sAcc[nt] = wmma_f16(qa, kb[nt], sAcc[nt]);
    }

    // row softmax: unconditional padded-bias loads keep EXEC full (no branching)
#pragma unroll
    for (int r = 0; r < 8; ++r) {
      int srow = 16 * mt + r + 8 * hi;
      const float* brow = bp + srow * 64;
      float p0 = sAcc[0][r] + brow[lr];
      float p1 = sAcc[1][r] + brow[16 + lr];
      float p2 = sAcc[2][r] + brow[32 + lr];
      float p3 = sAcc[3][r] + brow[48 + lr];
      float mx = fmaxf(fmaxf(p0, p1), fmaxf(p2, p3));
#pragma unroll
      for (int m = 8; m >= 1; m >>= 1) mx = fmaxf(mx, __shfl_xor(mx, m, 32));
      p0 = __expf(p0 - mx); p1 = __expf(p1 - mx); p2 = __expf(p2 - mx); p3 = __expf(p3 - mx);
      float sum = p0 + p1 + p2 + p3;
#pragma unroll
      for (int m = 8; m >= 1; m >>= 1) sum += __shfl_xor(sum, m, 32);
      float inv = 1.f / sum;
      int pr = r + 8 * hi;
      sP[pr][lr]      = (_Float16)(p0 * inv);
      sP[pr][16 + lr] = (_Float16)(p1 * inv);
      sP[pr][32 + lr] = (_Float16)(p2 * inv);
      sP[pr][48 + lr] = (_Float16)(p3 * inv);
    }

    // O strip = P(16x64) @ V(64x32): 2 K-steps x 2 d-tiles
    Frag pa0, pa1;
    pa0.h[0] = *(const v8h*)(&sP[lr][hi * 8]);
    pa0.h[1] = *(const v8h*)(&sP[lr][16 + hi * 8]);
    pa1.h[0] = *(const v8h*)(&sP[lr][32 + hi * 8]);
    pa1.h[1] = *(const v8h*)(&sP[lr][48 + hi * 8]);

    v8f oAcc[2];
#pragma unroll
    for (int dt = 0; dt < 2; ++dt) {
      v8f z = {0.f, 0.f, 0.f, 0.f, 0.f, 0.f, 0.f, 0.f};
      oAcc[dt] = z;
      Frag vb0, vb1;
      const _Float16* base = &sVt[16 * dt + lr][0];
      vb0.h[0] = *(const v8h*)(base + hi * 8);
      vb0.h[1] = *(const v8h*)(base + 16 + hi * 8);
      vb1.h[0] = *(const v8h*)(base + 32 + hi * 8);
      vb1.h[1] = *(const v8h*)(base + 48 + hi * 8);
      oAcc[dt] = wmma_f16(pa0, vb0, oAcc[dt]);
      oAcc[dt] = wmma_f16(pa1, vb1, oAcc[dt]);
    }

#pragma unroll
    for (int dt = 0; dt < 2; ++dt)
#pragma unroll
      for (int r = 0; r < 8; ++r) {
        int l = 16 * mt + r + 8 * hi;
        if (l < L_)
          Ao[((size_t)w * L_ + l) * C_ + h * HD_ + 16 * dt + lr] = (_Float16)oAcc[dt][r];
      }
  }
}

// ---------------- depthwise 3x3 + BN2 + GELU (NHWC) ----------------
__global__ void __launch_bounds__(256)
dwconv_kernel(const _Float16* __restrict__ H1, const float* __restrict__ dww,
              const float* __restrict__ ab2, _Float16* __restrict__ H2) {
  const int pix = blockIdx.x;  // b*HW + n
  const int bb = pix / HW_, n = pix - bb * HW_;
  const int y = n / 56, x = n - y * 56;
  for (int c = threadIdx.x; c < HID_; c += 256) {
    float acc = 0.f;
#pragma unroll
    for (int ky = -1; ky <= 1; ++ky)
#pragma unroll
      for (int kx = -1; kx <= 1; ++kx) {
        int yy = y + ky, xx = x + kx;
        if (yy >= 0 && yy < 56 && xx >= 0 && xx < 56)
          acc += (float)H1[((size_t)bb * HW_ + yy * 56 + xx) * HID_ + c] *
                 dww[c * 9 + (ky + 1) * 3 + (kx + 1)];
      }
    float t = acc * ab2[c] + ab2[2048 + c];
    H2[(size_t)pix * HID_ + c] = (_Float16)gelu_exact(t);
  }
}

// ---------------- host orchestration ----------------
extern "C" void kernel_launch(void* const* d_in, const int* in_sizes, int n_in,
                              void* d_out, int out_size, void* d_ws, size_t ws_size,
                              hipStream_t stream) {
  (void)in_sizes; (void)n_in; (void)out_size; (void)ws_size;

  const float* x      = (const float*)d_in[0];
  const float* qkv_w  = (const float*)d_in[1];
  const float* qkv_b  = (const float*)d_in[2];
  const float* proj_w = (const float*)d_in[3];
  const float* proj_b = (const float*)d_in[4];
  const float* rpb    = (const float*)d_in[5];
  const float* ln1_g  = (const float*)d_in[6];
  const float* ln1_b  = (const float*)d_in[7];
  const float* ln2_g  = (const float*)d_in[8];
  const float* ln2_b  = (const float*)d_in[9];
  const float* fc1_w  = (const float*)d_in[10];
  const float* fc1_b  = (const float*)d_in[11];
  const float* bn1_g  = (const float*)d_in[12];
  const float* bn1_b  = (const float*)d_in[13];
  const float* bn1_m  = (const float*)d_in[14];
  const float* bn1_v  = (const float*)d_in[15];
  const float* dw_w   = (const float*)d_in[16];
  const float* dw_b   = (const float*)d_in[17];
  const float* bn2_g  = (const float*)d_in[18];
  const float* bn2_b  = (const float*)d_in[19];
  const float* bn2_m  = (const float*)d_in[20];
  const float* bn2_v  = (const float*)d_in[21];
  const float* fc2_w  = (const float*)d_in[22];
  const float* fc2_b  = (const float*)d_in[23];
  const float* bn3_g  = (const float*)d_in[24];
  const float* bn3_b  = (const float*)d_in[25];
  const float* bn3_m  = (const float*)d_in[26];
  const float* bn3_v  = (const float*)d_in[27];
  const int* rel_index = (const int*)d_in[28];

  char* ws = (char*)d_ws;
  size_t off = 0;
  auto alloc = [&](size_t bytes) {
    off = (off + 255) & ~(size_t)255;
    size_t o = off; off += bytes; return o;
  };

  _Float16* wqkv16 = (_Float16*)(ws + alloc((size_t)1536 * 512 * 2));
  _Float16* wproj16 = (_Float16*)(ws + alloc((size_t)512 * 512 * 2));
  _Float16* wfc1_16 = (_Float16*)(ws + alloc((size_t)2048 * 512 * 2));
  _Float16* wfc2_16 = (_Float16*)(ws + alloc((size_t)512 * 2048 * 2));
  float* ab1 = (float*)(ws + alloc(4096 * 4));
  float* ab2 = (float*)(ws + alloc(4096 * 4));
  float* ab3 = (float*)(ws + alloc(1024 * 4));
  float* biasPad = (float*)(ws + alloc((size_t)NH_ * 64 * 64 * 4));
  _Float16* hA = (_Float16*)(ws + alloc((size_t)MTOT_ * C_ * 2));   // LN1 out; reused as LN2 out
  _Float16* qkvb = (_Float16*)(ws + alloc((size_t)3 * NWIN_ * NH_ * L_ * HD_ * 2 + 8192));
  _Float16* Ao = (_Float16*)(ws + alloc((size_t)MTOT_ * C_ * 2));   // contiguous after qkvb
  float* xs1 = (float*)(ws + alloc((size_t)MTOT_ * C_ * 4));
  _Float16* H1 = (_Float16*)(ws + alloc((size_t)MTOT_ * HID_ * 2));
  _Float16* H2 = qkvb;   // reuse qkvb+Ao region (dead by then), 102.77MB available
  _Float16* A2 = hA;

  // 1) prep (BN folds + padded RPE bias)
  prep_kernel<<<dim3((4608 + NH_ * 64 * 64 + 255) / 256), dim3(256), 0, stream>>>(
      fc1_b, bn1_g, bn1_b, bn1_m, bn1_v, dw_b, bn2_g, bn2_b, bn2_m, bn2_v,
      fc2_b, bn3_g, bn3_b, bn3_m, bn3_v, rpb, rel_index, ab1, ab2, ab3, biasPad);

  // 2) weight conversion to f16 (already [N,K] row-major = WMMA Bt layout)
  cvt_f32_f16<<<dim3(1024), dim3(256), 0, stream>>>(qkv_w, wqkv16, 1536 * 512);
  cvt_f32_f16<<<dim3(512), dim3(256), 0, stream>>>(proj_w, wproj16, 512 * 512);
  cvt_f32_f16<<<dim3(1024), dim3(256), 0, stream>>>(fc1_w, wfc1_16, 2048 * 512);
  cvt_f32_f16<<<dim3(1024), dim3(256), 0, stream>>>(fc2_w, wfc2_16, 512 * 2048);

  // 3) LN1 + window partition
  ln1_win_kernel<<<dim3(MTOT_ / 8), dim3(256), 0, stream>>>(x, ln1_g, ln1_b, hA);

  // 4) QKV GEMM (25088x1536x512) -> scattered q(scaled)/k/v
  gemm_wmma_epi<EPI_QKV><<<dim3(196, 12), dim3(256), 0, stream>>>(
      hA, wqkv16, MTOT_, 1536, 512, qkv_b, nullptr, nullptr, qkvb);

  // 5) windowed attention with RPE
  attn_kernel<<<dim3(NWIN_ * NH_), dim3(32), 0, stream>>>(qkvb, biasPad, Ao);

  // 6) out-proj GEMM + un-window + residual -> xs1 (f32, token-major)
  gemm_wmma_epi<EPI_PROJ_RES><<<dim3(196, 4), dim3(256), 0, stream>>>(
      Ao, wproj16, MTOT_, 512, 512, proj_b, nullptr, x, xs1);

  // 7) LN2
  ln2_kernel<<<dim3(MTOT_ / 8), dim3(256), 0, stream>>>(xs1, ln2_g, ln2_b, A2);

  // 8) fc1 GEMM + BN1 + GELU -> H1 (NHWC f16)
  gemm_wmma_epi<EPI_BN_GELU><<<dim3(196, 16), dim3(256), 0, stream>>>(
      A2, wfc1_16, MTOT_, 2048, 512, ab1, ab1 + 2048, nullptr, H1);

  // 9) depthwise 3x3 + BN2 + GELU -> H2
  dwconv_kernel<<<dim3(MTOT_), dim3(256), 0, stream>>>(H1, dw_w, ab2, H2);

  // 10) fc2 GEMM + BN3 + GELU + residual -> d_out (NCHW f32)
  gemm_wmma_epi<EPI_FC2_OUT><<<dim3(196, 4), dim3(256), 0, stream>>>(
      H2, wfc2_16, MTOT_, 512, 2048, ab3, ab3 + 512, xs1, d_out);
}
